// Attention_10548439679253
// MI455X (gfx1250) — matrix operand order repaired and checked by reading, not compiled, not run on hardware
//
#include <hip/hip_runtime.h>
#include <hip/hip_bf16.h>

typedef __attribute__((ext_vector_type(16))) __bf16 v16bf;
typedef __attribute__((ext_vector_type(8)))  float  v8f;

#define B_ 16
#define S_ 2048
#define D_ 512
#define LDS_STRIDE 40   // 32 + 8 pad: 80B rows keep 16B alignment, dodge bank conflicts

__constant__ const float kScale = 0.04419417382415922f; // 1/sqrt(512)

// ---- WMMA fragment loaders from LDS (bf16, 16x16x32) -----------------------
// A (16x32, MxK): lane l (0-15) row M=l, holds K = khalf*8 + 0..7 in f[0..7]
// and K = 16 + khalf*8 + 0..7 in f[8..15]   (ISA 7.12.2 16-bit A table)
__device__ __forceinline__ v16bf ldsFragA(const __bf16* base, int row, int khalf) {
  const __bf16* p0 = base + row * LDS_STRIDE + khalf * 8;
  v16bf f;
#pragma unroll
  for (int j = 0; j < 8; ++j) f[j] = p0[j];
#pragma unroll
  for (int j = 0; j < 8; ++j) f[j + 8] = p0[16 + j];
  return f;
}

// B (32x16, KxN) staged in LDS as Bt[n][k]: lane group khalf holds
// K = khalf*8 + j in f[0..7] and K = 16 + khalf*8 + j in f[8..15], N = lane%16
__device__ __forceinline__ v16bf ldsFragB(const __bf16* base, int n, int khalf) {
  const __bf16* p = base + n * LDS_STRIDE + khalf * 8;
  v16bf f;
#pragma unroll
  for (int j = 0; j < 8; ++j) f[j] = p[j];
#pragma unroll
  for (int j = 0; j < 8; ++j) f[j + 8] = p[16 + j];
  return f;
}

// ---- f32 -> bf16 conversion ------------------------------------------------
__global__ void cvt_bf16(const float* __restrict__ in, __bf16* __restrict__ out, int n) {
  int i = blockIdx.x * blockDim.x + threadIdx.x;
  if (i < n) out[i] = (__bf16)in[i];
}

// f32 [k][n] -> bf16 transposed [n][k]  (tiny 512x512, one-time cost in L2)
__global__ void cvt_w_t(const float* __restrict__ W, __bf16* __restrict__ Wt) {
  int i = blockIdx.x * blockDim.x + threadIdx.x;   // over D*D
  int n = i >> 9, k = i & (D_ - 1);
  Wt[i] = (__bf16)W[k * D_ + n];
}

// ---- QKV projection GEMM: O[z] = X(32768x512) @ W[z](512x512) --------------
// W is pre-transposed, so both tiles stage with 16B vector copies.
// block = 256 threads (8 waves), tile 128(M) x 128(N); wave = 64(M) x 32(N)
__global__ __launch_bounds__(256)
void proj_gemm(const __bf16* __restrict__ X, const __bf16* __restrict__ WtAll,
               __bf16* __restrict__ Oall) {
  __shared__ __bf16 As[128 * LDS_STRIDE];
  __shared__ __bf16 Bs[128 * LDS_STRIDE];

  const int z = blockIdx.z;
  const __bf16* Wt = WtAll + (size_t)z * D_ * D_;           // [n][k]
  __bf16* O = Oall + (size_t)z * (size_t)(B_ * S_) * D_;
  const float oscale = (z == 0) ? kScale : 1.0f;            // fold 1/sqrt(D) into Q

  const int n0 = blockIdx.x * 128;
  const int m0 = blockIdx.y * 128;
  const int tid = threadIdx.x;
  const int lane = tid & 31, wid = tid >> 5;
  const int wm = wid >> 2, wn = wid & 3;                    // 2x4 wave grid
  const int lrow = lane & 15, khalf = lane >> 4;

  const v8f vzero = {0.f, 0.f, 0.f, 0.f, 0.f, 0.f, 0.f, 0.f};
  v8f acc[4][2];
#pragma unroll
  for (int mt = 0; mt < 4; ++mt)
#pragma unroll
    for (int nt = 0; nt < 2; ++nt) acc[mt][nt] = vzero;

  for (int kk = 0; kk < D_; kk += 32) {
    // stage A (X rows) and Bt (Wt rows): 128 rows x 32 cols each, 16B copies
#pragma unroll
    for (int it = 0; it < 2; ++it) {
      int i = tid + it * 256;                               // 0..511
      int row = i >> 2, c8 = i & 3;
      *(uint4*)&As[row * LDS_STRIDE + c8 * 8] =
          *(const uint4*)&X[(size_t)(m0 + row) * D_ + kk + c8 * 8];
      *(uint4*)&Bs[row * LDS_STRIDE + c8 * 8] =
          *(const uint4*)&Wt[(size_t)(n0 + row) * D_ + kk + c8 * 8];
      if (kk + 32 < D_) {                                   // gfx1250 global_prefetch_b8
        __builtin_prefetch(&X[(size_t)(m0 + row) * D_ + kk + 32 + c8 * 8], 0, 1);
        __builtin_prefetch(&Wt[(size_t)(n0 + row) * D_ + kk + 32 + c8 * 8], 0, 1);
      }
    }
    __syncthreads();

    v16bf bfr[2];
#pragma unroll
    for (int nt = 0; nt < 2; ++nt)
      bfr[nt] = ldsFragB(Bs, wn * 32 + nt * 16 + lrow, khalf);
#pragma unroll
    for (int mt = 0; mt < 4; ++mt) {
      v16bf afr = ldsFragA(As, wm * 64 + mt * 16 + lrow, khalf);
#pragma unroll
      for (int nt = 0; nt < 2; ++nt)
        acc[mt][nt] = __builtin_amdgcn_wmma_f32_16x16x32_bf16(
            false, afr, false, bfr[nt], (short)0, acc[mt][nt], false, false);
    }
    __syncthreads();
  }

  // D layout: VGPR r -> M = r + 8*khalf, N = lane%16
#pragma unroll
  for (int mt = 0; mt < 4; ++mt)
#pragma unroll
    for (int nt = 0; nt < 2; ++nt)
#pragma unroll
      for (int r = 0; r < 8; ++r) {
        int row = m0 + wm * 64 + mt * 16 + r + khalf * 8;
        int col = n0 + wn * 32 + nt * 16 + lrow;
        O[(size_t)row * D_ + col] = (__bf16)(acc[mt][nt][r] * oscale);
      }
}

// ---- score pass: mode 0 => per-row (max, sumexp); mode 1 => w[b,k] ---------
// block = 128 q rows of one batch; wave = 16 q rows x 128 keys per block.
// Q fragments for the wave's 16 rows are held in registers (16 x v16bf).
__global__ __launch_bounds__(256)
void attn_scores(const __bf16* __restrict__ Q, const __bf16* __restrict__ K,
                 float* __restrict__ Mrow, float* __restrict__ Zrow,
                 float* __restrict__ Wsum, int mode) {
  __shared__ __bf16 Qs[128 * LDS_STRIDE];
  __shared__ __bf16 Ks[128 * LDS_STRIDE];
  __shared__ float swsum[128];

  const int b  = blockIdx.x;
  const int q0 = blockIdx.y * 128;
  const int tid = threadIdx.x;
  const int lane = tid & 31, wid = tid >> 5;
  const int lrow = lane & 15, khalf = lane >> 4;
  const int qr0 = wid * 16;

  const __bf16* Qb = Q + (size_t)b * S_ * D_;
  const __bf16* Kb = K + (size_t)b * S_ * D_;

  // ---- preload this wave's Q fragments for all 16 K-dim steps --------------
  v16bf qfrag[16];
#pragma unroll
  for (int kd16 = 0; kd16 < 16; ++kd16) {
#pragma unroll
    for (int it = 0; it < 2; ++it) {
      int i = tid + it * 256;
      int row = i >> 2, c8 = i & 3;
      *(uint4*)&Qs[row * LDS_STRIDE + c8 * 8] =
          *(const uint4*)&Qb[(size_t)(q0 + row) * D_ + kd16 * 32 + c8 * 8];
    }
    __syncthreads();
    qfrag[kd16] = ldsFragA(Qs, qr0 + lrow, khalf);
    __syncthreads();
  }

  float m_run[8], z_run[8], mrow[8], rz[8];
#pragma unroll
  for (int r = 0; r < 8; ++r) { m_run[r] = -3.0e38f; z_run[r] = 0.f; }
  if (mode == 1) {
#pragma unroll
    for (int r = 0; r < 8; ++r) {
      int row = q0 + qr0 + r + khalf * 8;
      mrow[r] = Mrow[b * S_ + row];
      rz[r]   = __frcp_rn(Zrow[b * S_ + row]);
    }
  }

  const v8f vzero = {0.f, 0.f, 0.f, 0.f, 0.f, 0.f, 0.f, 0.f};

  for (int kb = 0; kb < S_ / 128; ++kb) {
    v8f acc[8];
#pragma unroll
    for (int t = 0; t < 8; ++t) acc[t] = vzero;

#pragma unroll
    for (int kd16 = 0; kd16 < 16; ++kd16) {
#pragma unroll
      for (int it = 0; it < 2; ++it) {
        int i = tid + it * 256;
        int row = i >> 2, c8 = i & 3;
        *(uint4*)&Ks[row * LDS_STRIDE + c8 * 8] =
            *(const uint4*)&Kb[(size_t)(kb * 128 + row) * D_ + kd16 * 32 + c8 * 8];
        if (kd16 < 15)                                     // prefetch next chunk
          __builtin_prefetch(
              &Kb[(size_t)(kb * 128 + row) * D_ + (kd16 + 1) * 32 + c8 * 8], 0, 1);
      }
      __syncthreads();
      // S = Q @ K^T : Ks[key][d] is exactly the Bt[n][k] layout (n=key, k=d)
#pragma unroll
      for (int t = 0; t < 8; ++t) {
        v16bf bfr = ldsFragB(Ks, t * 16 + lrow, khalf);
        acc[t] = __builtin_amdgcn_wmma_f32_16x16x32_bf16(
            false, qfrag[kd16], false, bfr, (short)0, acc[t], false, false);
      }
      __syncthreads();
    }

    if (mode == 0) {
      // online softmax stats; a q-row lives across a 16-lane N group
#pragma unroll
      for (int r = 0; r < 8; ++r) {
        float tmax = acc[0][r];
#pragma unroll
        for (int t = 1; t < 8; ++t) tmax = fmaxf(tmax, acc[t][r]);
#pragma unroll
        for (int off = 8; off >= 1; off >>= 1)
          tmax = fmaxf(tmax, __shfl_xor(tmax, off, 16));
        float mnew = fmaxf(m_run[r], tmax);
        float es = 0.f;
#pragma unroll
        for (int t = 0; t < 8; ++t) es += __expf(acc[t][r] - mnew);
#pragma unroll
        for (int off = 8; off >= 1; off >>= 1) es += __shfl_xor(es, off, 16);
        z_run[r] = z_run[r] * __expf(m_run[r] - mnew) + es;
        m_run[r] = mnew;
      }
    } else {
      // column sums of attn probabilities -> w[b, key]
      if (tid < 128) swsum[tid] = 0.f;
      __syncthreads();
#pragma unroll
      for (int t = 0; t < 8; ++t) {
        float cs = 0.f;
#pragma unroll
        for (int r = 0; r < 8; ++r) cs += __expf(acc[t][r] - mrow[r]) * rz[r];
        cs += __shfl_xor(cs, 16, 32);     // combine M 0..7 with M 8..15 halves
        if (khalf == 0) atomicAdd(&swsum[t * 16 + lrow], cs);
      }
      __syncthreads();
      if (tid < 128) atomicAdd(&Wsum[b * S_ + kb * 128 + tid], swsum[tid]);
      __syncthreads();
    }
  }

  if (mode == 0 && lrow == 0) {
#pragma unroll
    for (int r = 0; r < 8; ++r) {
      int row = q0 + qr0 + r + khalf * 8;
      Mrow[b * S_ + row] = m_run[r];
      Zrow[b * S_ + row] = z_run[r];
    }
  }
}

// ---- out[b,d] = sum_k w[b,k] * V[b,k,d] ------------------------------------
__global__ __launch_bounds__(256)
void weighted_v(const __bf16* __restrict__ V, const float* __restrict__ Wsum,
                float* __restrict__ out) {
  const int b = blockIdx.y;
  const int d = blockIdx.x * 256 + threadIdx.x;
  const __bf16* Vb = V + (size_t)b * S_ * D_;
  const float* wb = Wsum + b * S_;
  float acc = 0.f;
  for (int k = 0; k < S_; ++k)
    acc = fmaf(wb[k], (float)Vb[(size_t)k * D_ + d], acc);
  out[b * D_ + d] = acc;
}

extern "C" void kernel_launch(void* const* d_in, const int* in_sizes, int n_in,
                              void* d_out, int out_size, void* d_ws, size_t ws_size,
                              hipStream_t stream) {
  (void)in_sizes; (void)n_in; (void)out_size; (void)ws_size;
  const float* x  = (const float*)d_in[0];
  const float* wq = (const float*)d_in[1];
  const float* wk = (const float*)d_in[2];
  const float* wv = (const float*)d_in[3];
  float* out = (float*)d_out;

  const size_t BSD = (size_t)B_ * S_ * D_;   // 16,777,216 elems
  const size_t DD  = (size_t)D_ * D_;        // 262,144 elems

  char* ws = (char*)d_ws;
  size_t off = 0;
  auto alloc = [&](size_t bytes) -> char* {
    char* p = ws + off;
    off = (off + bytes + 255) & ~(size_t)255;
    return p;
  };
  __bf16* Xb   = (__bf16*)alloc(BSD * 2);          //  32 MB
  __bf16* Wtb  = (__bf16*)alloc(3 * DD * 2);       // 1.5 MB (Wq^T, Wk^T, Wv^T)
  __bf16* QKV  = (__bf16*)alloc(3 * BSD * 2);      //  96 MB (contiguous Q,K,V)
  float*  Mrow = (float*)alloc((size_t)B_ * S_ * 4);
  float*  Zrow = (float*)alloc((size_t)B_ * S_ * 4);
  float*  Wsum = (float*)alloc((size_t)B_ * S_ * 4);
  __bf16* Qb = QKV;
  __bf16* Kb = QKV + BSD;
  __bf16* Vb = QKV + 2 * BSD;

  // 1) precision conversion; weights transposed so GEMM B-tiles stage vectorized
  cvt_bf16<<<(int)((BSD + 255) / 256), 256, 0, stream>>>(x, Xb, (int)BSD);
  cvt_w_t<<<(int)(DD / 256), 256, 0, stream>>>(wq, Wtb);
  cvt_w_t<<<(int)(DD / 256), 256, 0, stream>>>(wk, Wtb + DD);
  cvt_w_t<<<(int)(DD / 256), 256, 0, stream>>>(wv, Wtb + 2 * DD);

  // 2) Q/K/V projections: grid (N tiles, M tiles, 3 weights)
  proj_gemm<<<dim3(D_ / 128, (B_ * S_) / 128, 3), 256, 0, stream>>>(Xb, Wtb, QKV);

  // 3) softmax stats pass
  attn_scores<<<dim3(B_, S_ / 128), 256, 0, stream>>>(Qb, Kb, Mrow, Zrow, Wsum, 0);

  // 4) per-key probability column sums
  hipMemsetAsync(Wsum, 0, (size_t)B_ * S_ * 4, stream);
  attn_scores<<<dim3(B_, S_ / 128), 256, 0, stream>>>(Qb, Kb, Mrow, Zrow, Wsum, 1);

  // 5) weighted reduction over V -> [B, D]
  weighted_v<<<dim3(D_ / 256, B_), 256, 0, stream>>>(Vb, Wsum, out);
}
